// Model_37194416783929
// MI455X (gfx1250) — matrix operand order repaired and checked by reading, not compiled, not run on hardware
//
#include <hip/hip_runtime.h>

// ---------------------------------------------------------------------------
// CDNA5 (gfx1250) implementation: RNN -> GATv2 -> MHA decoder -> MLP head.
// Dense contractions: v_wmma_f32_16x16x32_bf16 (wave32 WMMA).
// Common-case GEMMs stage the B panel into LDS with the Tensor Data Mover
// (tensor_load_to_lds + s_wait_tensorcnt), then gather fragments from LDS.
// ---------------------------------------------------------------------------

typedef __bf16 bf16_t;
typedef __attribute__((ext_vector_type(16))) __bf16 v16bf;
typedef __attribute__((ext_vector_type(8)))  float  v8f;
typedef __attribute__((ext_vector_type(4)))  unsigned int v4u;
typedef __attribute__((ext_vector_type(8)))  int v8i;
typedef __attribute__((ext_vector_type(4)))  int v4i;

#define B_     32
#define WSZ    128
#define FN     256
#define EMB    256
#define GH     2
#define E0     4096
#define NNODE  (B_ * FN)            // 8192
#define NEDGE  (B_ * E0 + NNODE)    // 139264
#define BH     64                   // z batch
#define AH     4                    // attn heads
#define DH     64                   // head dim

// ============================ WMMA fragment loaders =========================
// 16-bit A matrix 16x32 (ISA 7.12.2): lanes 0-15 rows M=0-15.
//   VGPR v : K = half*16 + 2v {,+1} here, not the ISA table order (matches B fragment order)
__device__ __forceinline__ v16bf load_frag_a(const bf16_t* base, int lane,
                                             long sM, long sK) {
    int half = lane >> 4, m = lane & 15;
    v16bf r;
#pragma unroll
    for (int v = 0; v < 8; ++v) {
        int kb = half * 16 + 2 * v;
        r[2 * v]     = base[(long)m * sM + (long)kb * sK];
        r[2 * v + 1] = base[(long)m * sM + (long)(kb + 1) * sK];
    }
    return r;
}
// 16-bit B matrix 32x16: lanes 0-15 hold K=0-15 (VGPR v: K=2v,2v+1),
// lanes 16-31 hold K=16-31.
__device__ __forceinline__ v16bf load_frag_b(const bf16_t* base, int lane,
                                             long sK, long sN) {
    int n = lane & 15, kh = (lane >> 4) * 16;
    v16bf r;
#pragma unroll
    for (int v = 0; v < 8; ++v) {
        int k = kh + 2 * v;
        r[2 * v]     = base[(long)k * sK + (long)n * sN];
        r[2 * v + 1] = base[(long)(k + 1) * sK + (long)n * sN];
    }
    return r;
}

// =========================== Generic WMMA GEMM ==============================
// C[bz] = act( scale * A[bz] x B[bz] + bias ),  one 16x16 tile per wave.
// Used only for the q@k^T logits GEMM (B column-major gather).
__global__ void gemm_bf16(const bf16_t* __restrict__ A, long aBO, long aBI,
                          long aM, long aK,
                          const bf16_t* __restrict__ B, long bBO, long bBI,
                          long bK, long bN,
                          float* __restrict__ C, long cBO, long cBI,
                          long cM, long cN,
                          const float* __restrict__ bias, int act, float scale,
                          int K, int inner) {
    int wave = threadIdx.x >> 5;
    int lane = threadIdx.x & 31;
    int tm = blockIdx.x;
    int tn = blockIdx.y * 4 + wave;
    int bz = blockIdx.z;
    int outer = bz / inner, in = bz % inner;
    const bf16_t* Ab = A + outer * aBO + in * aBI + (long)tm * 16 * aM;
    const bf16_t* Bb = B + outer * bBO + in * bBI + (long)tn * 16 * bN;
    float*        Cb = C + outer * cBO + in * cBI + (long)tm * 16 * cM
                         + (long)tn * 16 * cN;
    v8f acc = {};
    for (int k0 = 0; k0 < K; k0 += 32) {
        v16bf a = load_frag_a(Ab + (long)k0 * aK, lane, aM, aK);
        v16bf b = load_frag_b(Bb + (long)k0 * bK, lane, bK, bN);
        acc = __builtin_amdgcn_wmma_f32_16x16x32_bf16(
            false, a, false, b, (short)0, acc, false, false);
    }
    int half = lane >> 4, n = lane & 15;
#pragma unroll
    for (int v = 0; v < 8; ++v) {
        int m = v + half * 8;
        float x = acc[v] * scale;
        if (bias) x += bias[tn * 16 + n];
        if (act == 1) x = fmaxf(x, 0.f);
        else if (act == 2) x = tanhf(x);
        Cb[(long)m * cM + (long)n * cN] = x;
    }
}

// ===================== TDM-staged WMMA GEMM (B row-major) ===================
// Block = 4 waves = one 16(M) x 64(N) tile. The [K x 64] bf16 B panel is
// DMA'd into LDS once by the Tensor Data Mover, then all waves gather
// fragments from LDS. A fragments come from global (arbitrary strides).
__global__ void gemm_bf16_tdm(const bf16_t* __restrict__ A, long aBO, long aBI,
                              long aM, long aK,
                              const bf16_t* __restrict__ B, long bBO, long bBI,
                              long bRow, int Ntot,
                              float* __restrict__ C, long cBO, long cBI,
                              long cM, long cN,
                              const float* __restrict__ bias, int act,
                              float scale, int K, int inner) {
    __shared__ bf16_t sB[256 * 64];            // up to K=256 panel (32 KB)
    int wave = threadIdx.x >> 5;
    int lane = threadIdx.x & 31;
    int tm = blockIdx.x;
    int tnb = blockIdx.y;                      // 64-wide N block
    int bz = blockIdx.z;
    int outer = bz / inner, in = bz % inner;
    const bf16_t* Bb = B + outer * bBO + in * bBI + (long)tnb * 64;

    if (threadIdx.x < 32) {
        // ---- Tensor DMA descriptor (D#), CDNA5 ISA ch.8 ----
        unsigned lds = (unsigned)(uintptr_t)sB;           // LDS byte offset
        unsigned long long ga = (unsigned long long)(uintptr_t)Bb;
        v4u g0 = { 1u,                                    // count=1, user D#
                   lds,                                   // lds_addr
                   (unsigned)(ga & 0xffffffffu),          // global_addr[31:0]
                   (unsigned)((ga >> 32) & 0x01ffffffu)   // global_addr[56:32]
                   | (2u << 30) };                        // type=2 ("image")
        unsigned long long rs = (unsigned long long)bRow; // row stride (elems)
        v8i g1;
        g1[0] = 0x00010000;                               // data_size=2B
        g1[1] = (int)(((unsigned)Ntot & 0xffffu) << 16);  // tensor_dim0 lo
        g1[2] = (int)((((unsigned)Ntot >> 16) & 0xffffu)
                    | (((unsigned)K & 0xffffu) << 16));   // dim0 hi | dim1 lo
        g1[3] = (int)((((unsigned)K >> 16) & 0xffffu)
                    | (64u << 16));                       // dim1 hi | tile_dim0=64
        g1[4] = (int)((unsigned)K & 0xffffu);             // tile_dim1=K rows
        g1[5] = (int)(rs & 0xffffffffu);                  // dim0_stride lo
        g1[6] = (int)((rs >> 32) & 0xffffu);              // dim0_stride hi
        g1[7] = 0;
        v4i g2 = {0, 0, 0, 0};
        v4i g3 = {0, 0, 0, 0};
        v8i g4 = {0, 0, 0, 0, 0, 0, 0, 0};
        __builtin_amdgcn_tensor_load_to_lds(g0, g1, g2, g3, g4, 0);
        __builtin_amdgcn_s_wait_tensorcnt(0);
    }
    __syncthreads();

    const bf16_t* Ab = A + outer * aBO + in * aBI + (long)tm * 16 * aM;
    v8f acc = {};
    for (int k0 = 0; k0 < K; k0 += 32) {
        v16bf a = load_frag_a(Ab + (long)k0 * aK, lane, aM, aK);
        v16bf b = load_frag_b(sB + (long)k0 * 64 + wave * 16, lane, 64, 1);
        acc = __builtin_amdgcn_wmma_f32_16x16x32_bf16(
            false, a, false, b, (short)0, acc, false, false);
    }
    int half = lane >> 4, n = lane & 15;
    int ncol = tnb * 64 + wave * 16 + n;
    float* Cb = C + outer * cBO + in * cBI + (long)tm * 16 * cM;
#pragma unroll
    for (int v = 0; v < 8; ++v) {
        int m = v + half * 8;
        float x = acc[v] * scale;
        if (bias) x += bias[ncol];
        if (act == 1) x = fmaxf(x, 0.f);
        else if (act == 2) x = tanhf(x);
        Cb[(long)m * cM + (long)ncol * cN] = x;
    }
}

// =============== Fused RNN step: h' = tanh(x_t Wih^T + h Whh^T + b) =========
__global__ void rnn_step(const bf16_t* __restrict__ xT,
                         const bf16_t* __restrict__ WihT,
                         const bf16_t* __restrict__ h,
                         const bf16_t* __restrict__ WhhT,
                         const float* __restrict__ bih,
                         const float* __restrict__ bhh,
                         float* __restrict__ y, bf16_t* __restrict__ hN) {
    int wave = threadIdx.x >> 5;
    int lane = threadIdx.x & 31;
    int tm = blockIdx.x;
    int tn = blockIdx.y * 4 + wave;
    const bf16_t* Ax = xT + (long)tm * 16 * WSZ;
    const bf16_t* Ah = h  + (long)tm * 16 * EMB;
    v8f acc = {};
#pragma unroll
    for (int k0 = 0; k0 < WSZ; k0 += 32) {
        v16bf a = load_frag_a(Ax + k0, lane, WSZ, 1);
        v16bf b = load_frag_b(WihT + (long)k0 * EMB + tn * 16, lane, EMB, 1);
        acc = __builtin_amdgcn_wmma_f32_16x16x32_bf16(
            false, a, false, b, (short)0, acc, false, false);
    }
#pragma unroll
    for (int k0 = 0; k0 < EMB; k0 += 32) {
        v16bf a = load_frag_a(Ah + k0, lane, EMB, 1);
        v16bf b = load_frag_b(WhhT + (long)k0 * EMB + tn * 16, lane, EMB, 1);
        acc = __builtin_amdgcn_wmma_f32_16x16x32_bf16(
            false, a, false, b, (short)0, acc, false, false);
    }
    int half = lane >> 4, n = lane & 15;
#pragma unroll
    for (int v = 0; v < 8; ++v) {
        int m = tm * 16 + v + half * 8;
        int col = tn * 16 + n;
        float x = tanhf(acc[v] + bih[col] + bhh[col]);
        y[(long)m * EMB + col]  = x;
        hN[(long)m * EMB + col] = (bf16_t)x;
    }
}

// ============================ Elementwise helpers ===========================
__global__ void k_f32_to_bf16(const float* __restrict__ s,
                              bf16_t* __restrict__ d, long n) {
    long i = (long)blockIdx.x * blockDim.x + threadIdx.x;
    long stride = (long)gridDim.x * blockDim.x;
    for (; i < n; i += stride) d[i] = (bf16_t)s[i];
}
// dst[b][c][r] = src[b][r][c]
__global__ void k_transpose_bf16(const float* __restrict__ s,
                                 bf16_t* __restrict__ d, int R, int C, long n) {
    long i = (long)blockIdx.x * blockDim.x + threadIdx.x;
    long stride = (long)gridDim.x * blockDim.x;
    long rc = (long)R * C;
    for (; i < n; i += stride) {
        long b = i / rc, rem = i - b * rc;
        long r = rem / C, c = rem - r * C;
        d[b * rc + c * R + r] = (bf16_t)s[i];
    }
}
__global__ void k_zbias_bf16(const float* __restrict__ aggr,
                             const float* __restrict__ gbias,
                             bf16_t* __restrict__ z, long n) {
    long i = (long)blockIdx.x * blockDim.x + threadIdx.x;
    long stride = (long)gridDim.x * blockDim.x;
    for (; i < n; i += stride) z[i] = (bf16_t)(aggr[i] + gbias[i & 511]);
}

// ================================ Edge stage ================================
__global__ void k_build_edges(const int* __restrict__ eidx,
                              int* __restrict__ srcE, int* __restrict__ dstE) {
    int i = blockIdx.x * blockDim.x + threadIdx.x;
    const int NE = B_ * E0;
    if (i < NE) {
        int g = i >> 12, j = i & (E0 - 1);
        srcE[i] = eidx[j] + g * EMB;
        dstE[i] = eidx[E0 + j] + g * EMB;
    } else if (i < NEDGE) {
        int v = i - NE;
        srcE[i] = v; dstE[i] = v;
    }
}
__global__ void k_edge_score(const float* __restrict__ xl,
                             const float* __restrict__ xr,
                             const float* __restrict__ att,
                             const int* __restrict__ srcE,
                             const int* __restrict__ dstE,
                             float* __restrict__ score) {
    int wid = (int)(((long)blockIdx.x * blockDim.x + threadIdx.x) >> 5);
    int lane = threadIdx.x & 31;
    if (wid >= NEDGE) return;
    long s = (long)srcE[wid] * (GH * EMB);
    long d = (long)dstE[wid] * (GH * EMB);
    float a0 = 0.f, a1 = 0.f;
    for (int c = lane; c < EMB; c += 32) {
        float e0 = xl[s + c] + xr[d + c];
        e0 = e0 > 0.f ? e0 : 0.2f * e0;
        a0 += att[c] * e0;
        float e1 = xl[s + EMB + c] + xr[d + EMB + c];
        e1 = e1 > 0.f ? e1 : 0.2f * e1;
        a1 += att[EMB + c] * e1;
    }
    for (int off = 16; off; off >>= 1) {
        a0 += __shfl_down(a0, off, 32);
        a1 += __shfl_down(a1, off, 32);
    }
    if (lane == 0) { score[wid * 2] = a0; score[wid * 2 + 1] = a1; }
}
__device__ __forceinline__ unsigned f2key(float f) {
    unsigned b = __float_as_uint(f);
    return (b & 0x80000000u) ? ~b : (b | 0x80000000u);
}
__device__ __forceinline__ float key2f(unsigned k) {
    return __uint_as_float((k & 0x80000000u) ? (k & 0x7fffffffu) : ~k);
}
__global__ void k_seg_max(const float* __restrict__ score,
                          const int* __restrict__ dstE,
                          unsigned* __restrict__ mkey) {
    int i = blockIdx.x * blockDim.x + threadIdx.x;
    if (i >= NEDGE * GH) return;
    int e = i >> 1, h = i & 1;
    atomicMax(&mkey[dstE[e] * GH + h], f2key(score[i]));
}
__global__ void k_edge_exp(const float* __restrict__ score,
                           const int* __restrict__ dstE,
                           const unsigned* __restrict__ mkey,
                           float* __restrict__ exb, float* __restrict__ den) {
    int i = blockIdx.x * blockDim.x + threadIdx.x;
    if (i >= NEDGE * GH) return;
    int e = i >> 1, h = i & 1;
    int seg = dstE[e] * GH + h;
    float ex = __expf(score[i] - key2f(mkey[seg]));
    exb[i] = ex;
    atomicAdd(&den[seg], ex);
}
__global__ void k_alpha(const float* __restrict__ exb,
                        const int* __restrict__ dstE,
                        const float* __restrict__ den,
                        float* __restrict__ alpha_ws,
                        float* __restrict__ alpha_out) {
    int i = blockIdx.x * blockDim.x + threadIdx.x;
    if (i >= NEDGE * GH) return;
    int e = i >> 1, h = i & 1;
    float a = exb[i] / den[dstE[e] * GH + h];
    alpha_ws[i] = a;
    alpha_out[i] = a;
}
__global__ void k_scatter(const float* __restrict__ xl,
                          const int* __restrict__ srcE,
                          const int* __restrict__ dstE,
                          const float* __restrict__ alpha,
                          float* __restrict__ aggr) {
    int wid = (int)(((long)blockIdx.x * blockDim.x + threadIdx.x) >> 5);
    int lane = threadIdx.x & 31;
    if (wid >= NEDGE) return;
    long s = (long)srcE[wid] * (GH * EMB);
    long d = (long)dstE[wid] * (GH * EMB);
    float a0 = alpha[wid * 2], a1 = alpha[wid * 2 + 1];
    for (int idx = lane; idx < GH * EMB; idx += 32) {
        float a = (idx < EMB) ? a0 : a1;
        atomicAdd(&aggr[d + idx], a * xl[s + idx]);
    }
}

// ========================= Row softmax -> bf16 ==============================
__global__ void k_softmax_row(const float* __restrict__ logits,
                              bf16_t* __restrict__ w, long rows) {
    long wid = ((long)blockIdx.x * blockDim.x + threadIdx.x) >> 5;
    int lane = threadIdx.x & 31;
    if (wid >= rows) return;
    const float* r = logits + wid * 256;
    float vals[8], mx = -3.4e38f;
#pragma unroll
    for (int i = 0; i < 8; ++i) { vals[i] = r[lane + 32 * i]; mx = fmaxf(mx, vals[i]); }
    for (int off = 16; off; off >>= 1) mx = fmaxf(mx, __shfl_xor(mx, off, 32));
    float sum = 0.f;
#pragma unroll
    for (int i = 0; i < 8; ++i) { vals[i] = __expf(vals[i] - mx); sum += vals[i]; }
    for (int off = 16; off; off >>= 1) sum += __shfl_xor(sum, off, 32);
    float inv = 1.f / sum;
#pragma unroll
    for (int i = 0; i < 8; ++i) w[wid * 256 + lane + 32 * i] = (bf16_t)(vals[i] * inv);
}

// ======================== Forecast head (N=1 GEMV) ==========================
__global__ void k_forecast(const float* __restrict__ h1,
                           const float* __restrict__ W2,
                           const float* __restrict__ b2,
                           float* __restrict__ out, long rows) {
    long wid = ((long)blockIdx.x * blockDim.x + threadIdx.x) >> 5;
    int lane = threadIdx.x & 31;
    if (wid >= rows) return;
    float a = 0.f;
    for (int e = lane; e < EMB; e += 32) a += h1[wid * EMB + e] * W2[e];
    for (int off = 16; off; off >>= 1) a += __shfl_down(a, off, 32);
    if (lane == 0) out[wid] = a + b2[0];
}

// ================================ Host side =================================
static inline void gemm(hipStream_t st,
                        const bf16_t* A, long aBO, long aBI, long aM, long aK,
                        const bf16_t* B, long bBO, long bBI, long bK, long bN,
                        float* C, long cBO, long cBI, long cM, long cN,
                        const float* bias, int act, float scale,
                        int M, int N, int K, int batches, int inner) {
    dim3 grid(M / 16, N / 64, batches);
    gemm_bf16<<<grid, 128, 0, st>>>(A, aBO, aBI, aM, aK, B, bBO, bBI, bK, bN,
                                    C, cBO, cBI, cM, cN, bias, act, scale,
                                    K, inner);
}
static inline void gemmT(hipStream_t st,
                         const bf16_t* A, long aBO, long aBI, long aM, long aK,
                         const bf16_t* B, long bBO, long bBI, long bRow, int Ntot,
                         float* C, long cBO, long cBI, long cM, long cN,
                         const float* bias, int act, float scale,
                         int M, int N, int K, int batches, int inner) {
    dim3 grid(M / 16, N / 64, batches);
    gemm_bf16_tdm<<<grid, 128, 0, st>>>(A, aBO, aBI, aM, aK,
                                        B, bBO, bBI, bRow, Ntot,
                                        C, cBO, cBI, cM, cN, bias, act, scale,
                                        K, inner);
}

extern "C" void kernel_launch(void* const* d_in, const int* in_sizes, int n_in,
                              void* d_out, int out_size, void* d_ws, size_t ws_size,
                              hipStream_t stream) {
    const float* x       = (const float*)d_in[0];
    const int*   eidx    = (const int*)  d_in[1];
    const float* W_ih    = (const float*)d_in[2];
    const float* b_ih    = (const float*)d_in[3];
    const float* W_hh    = (const float*)d_in[4];
    const float* b_hh    = (const float*)d_in[5];
    const float* Wl      = (const float*)d_in[6];
    const float* Wr      = (const float*)d_in[7];
    const float* att     = (const float*)d_in[8];
    const float* gbias   = (const float*)d_in[9];
    const float* Wq      = (const float*)d_in[10];
    const float* bq      = (const float*)d_in[11];
    const float* Wk      = (const float*)d_in[12];
    const float* bk      = (const float*)d_in[13];
    const float* Wv      = (const float*)d_in[14];
    const float* bv      = (const float*)d_in[15];
    const float* Wo      = (const float*)d_in[16];
    const float* bo      = (const float*)d_in[17];
    const float* W1      = (const float*)d_in[18];
    const float* b1      = (const float*)d_in[19];
    const float* W2      = (const float*)d_in[20];
    const float* b2      = (const float*)d_in[21];
    float* out = (float*)d_out;

    char* base = (char*)d_ws;
    size_t off = 0;
    auto alloc = [&](size_t bytes) -> char* {
        char* p = base + off;
        off = (off + bytes + 255) & ~(size_t)255;
        return p;
    };
    bf16_t* xT    = (bf16_t*)alloc((size_t)B_ * FN * WSZ * 2);
    bf16_t* WihT  = (bf16_t*)alloc((size_t)WSZ * EMB * 2);
    bf16_t* WhhT  = (bf16_t*)alloc((size_t)EMB * EMB * 2);
    bf16_t* Wl_b  = (bf16_t*)alloc((size_t)EMB * GH * EMB * 2);
    bf16_t* Wr_b  = (bf16_t*)alloc((size_t)EMB * GH * EMB * 2);
    bf16_t* Wq_b  = (bf16_t*)alloc((size_t)FN * FN * 2);
    bf16_t* Wk_b  = (bf16_t*)alloc((size_t)FN * FN * 2);
    bf16_t* Wv_b  = (bf16_t*)alloc((size_t)FN * FN * 2);
    bf16_t* Wo_b  = (bf16_t*)alloc((size_t)FN * FN * 2);
    bf16_t* W1_b  = (bf16_t*)alloc((size_t)EMB * EMB * 2);
    bf16_t* hA    = (bf16_t*)alloc((size_t)FN * EMB * 2);
    bf16_t* hB    = (bf16_t*)alloc((size_t)FN * EMB * 2);
    float*  nodes = (float*) alloc((size_t)NNODE * EMB * 4);
    bf16_t* nodes_b = (bf16_t*)alloc((size_t)NNODE * EMB * 2);
    float*  xl    = (float*) alloc((size_t)NNODE * GH * EMB * 4);
    float*  xr    = (float*) alloc((size_t)NNODE * GH * EMB * 4);
    int*    srcE  = (int*)   alloc((size_t)NEDGE * 4);
    int*    dstE  = (int*)   alloc((size_t)NEDGE * 4);
    float*  score = (float*) alloc((size_t)NEDGE * GH * 4);
    unsigned* mkey= (unsigned*)alloc((size_t)NNODE * GH * 4);
    float*  den   = (float*) alloc((size_t)NNODE * GH * 4);
    float*  exb   = (float*) alloc((size_t)NEDGE * GH * 4);
    float*  alpha = (float*) alloc((size_t)NEDGE * GH * 4);
    float*  aggr  = (float*) alloc((size_t)NNODE * GH * EMB * 4);
    bf16_t* z_b   = (bf16_t*)alloc((size_t)BH * EMB * FN * 2);
    float*  scrF  = (float*) alloc((size_t)BH * EMB * FN * 4);
    bf16_t* q_b   = (bf16_t*)alloc((size_t)BH * EMB * FN * 2);
    bf16_t* k_b   = (bf16_t*)alloc((size_t)BH * EMB * FN * 2);
    bf16_t* v_b   = (bf16_t*)alloc((size_t)BH * EMB * FN * 2);
    float*  logit = (float*) alloc((size_t)BH * AH * EMB * EMB * 4);
    bf16_t* w_b   = (bf16_t*)alloc((size_t)BH * AH * EMB * EMB * 2);
    bf16_t* at_b  = (bf16_t*)alloc((size_t)BH * EMB * FN * 2);
    (void)ws_size; (void)n_in; (void)in_sizes; (void)out_size;

    float* recon_out = out;
    float* fc_out    = out + (size_t)BH * EMB * FN;
    float* alpha_out = fc_out + (size_t)BH * FN;

    const int TPB = 256;
    auto blks = [](long n, int t) { return (unsigned)((n + t - 1) / t); };

    // ---- weight / input conversion ----
    k_transpose_bf16<<<blks((long)EMB * WSZ, TPB), TPB, 0, stream>>>(
        W_ih, WihT, EMB, WSZ, (long)EMB * WSZ);
    k_transpose_bf16<<<blks((long)EMB * EMB, TPB), TPB, 0, stream>>>(
        W_hh, WhhT, EMB, EMB, (long)EMB * EMB);
    k_transpose_bf16<<<blks((long)B_ * WSZ * FN, TPB), TPB, 0, stream>>>(
        x, xT, WSZ, FN, (long)B_ * WSZ * FN);
    k_f32_to_bf16<<<blks((long)EMB * GH * EMB, TPB), TPB, 0, stream>>>(Wl, Wl_b, (long)EMB * GH * EMB);
    k_f32_to_bf16<<<blks((long)EMB * GH * EMB, TPB), TPB, 0, stream>>>(Wr, Wr_b, (long)EMB * GH * EMB);
    k_f32_to_bf16<<<blks((long)FN * FN, TPB), TPB, 0, stream>>>(Wq, Wq_b, (long)FN * FN);
    k_f32_to_bf16<<<blks((long)FN * FN, TPB), TPB, 0, stream>>>(Wk, Wk_b, (long)FN * FN);
    k_f32_to_bf16<<<blks((long)FN * FN, TPB), TPB, 0, stream>>>(Wv, Wv_b, (long)FN * FN);
    k_f32_to_bf16<<<blks((long)FN * FN, TPB), TPB, 0, stream>>>(Wo, Wo_b, (long)FN * FN);
    k_f32_to_bf16<<<blks((long)EMB * EMB, TPB), TPB, 0, stream>>>(W1, W1_b, (long)EMB * EMB);

    // ---- zero inits (graph replays must be idempotent) ----
    (void)hipMemsetAsync(hA,   0, (size_t)FN * EMB * 2, stream);
    (void)hipMemsetAsync(mkey, 0, (size_t)NNODE * GH * 4, stream);
    (void)hipMemsetAsync(den,  0, (size_t)NNODE * GH * 4, stream);
    (void)hipMemsetAsync(aggr, 0, (size_t)NNODE * GH * EMB * 4, stream);

    // ---- RNN scan over B (32 sequential WMMA steps) ----
    bf16_t* hp = hA; bf16_t* hn = hB;
    for (int t = 0; t < B_; ++t) {
        rnn_step<<<dim3(FN / 16, EMB / 64), 128, 0, stream>>>(
            xT + (long)t * FN * WSZ, WihT, hp, WhhT, b_ih, b_hh,
            nodes + (long)t * FN * EMB, hn);
        bf16_t* tmp = hp; hp = hn; hn = tmp;
    }
    k_f32_to_bf16<<<blks((long)NNODE * EMB, TPB), TPB, 0, stream>>>(
        nodes, nodes_b, (long)NNODE * EMB);

    // ---- GATv2 projections (TDM-staged B panels) ----
    gemmT(stream, nodes_b, 0, 0, EMB, 1, Wl_b, 0, 0, GH * EMB, GH * EMB,
          xl, 0, 0, GH * EMB, 1, nullptr, 0, 1.f, NNODE, GH * EMB, EMB, 1, 1);
    gemmT(stream, nodes_b, 0, 0, EMB, 1, Wr_b, 0, 0, GH * EMB, GH * EMB,
          xr, 0, 0, GH * EMB, 1, nullptr, 0, 1.f, NNODE, GH * EMB, EMB, 1, 1);

    // ---- edges + segment softmax + scatter ----
    k_build_edges<<<blks(NEDGE, TPB), TPB, 0, stream>>>(eidx, srcE, dstE);
    k_edge_score<<<blks((long)NEDGE * 32, TPB), TPB, 0, stream>>>(
        xl, xr, att, srcE, dstE, score);
    k_seg_max<<<blks((long)NEDGE * GH, TPB), TPB, 0, stream>>>(score, dstE, mkey);
    k_edge_exp<<<blks((long)NEDGE * GH, TPB), TPB, 0, stream>>>(
        score, dstE, mkey, exb, den);
    k_alpha<<<blks((long)NEDGE * GH, TPB), TPB, 0, stream>>>(
        exb, dstE, den, alpha, alpha_out);
    k_scatter<<<blks((long)NEDGE * 32, TPB), TPB, 0, stream>>>(
        xl, srcE, dstE, alpha, aggr);
    k_zbias_bf16<<<blks((long)NNODE * GH * EMB, TPB), TPB, 0, stream>>>(
        aggr, gbias, z_b, (long)NNODE * GH * EMB);

    // ---- attention decoder ----
    const long ZB = (long)EMB * FN;  // 65536 per z batch
    gemmT(stream, z_b, 0, 0, FN, 1, Wq_b, 0, 0, FN, FN, scrF, 0, 0, FN, 1,
          bq, 0, 1.f, BH * EMB, FN, FN, 1, 1);
    k_f32_to_bf16<<<blks(BH * ZB, TPB), TPB, 0, stream>>>(scrF, q_b, BH * ZB);
    gemmT(stream, z_b, 0, 0, FN, 1, Wk_b, 0, 0, FN, FN, scrF, 0, 0, FN, 1,
          bk, 0, 1.f, BH * EMB, FN, FN, 1, 1);
    k_f32_to_bf16<<<blks(BH * ZB, TPB), TPB, 0, stream>>>(scrF, k_b, BH * ZB);
    gemmT(stream, z_b, 0, 0, FN, 1, Wv_b, 0, 0, FN, FN, scrF, 0, 0, FN, 1,
          bv, 0, 1.f, BH * EMB, FN, FN, 1, 1);
    k_f32_to_bf16<<<blks(BH * ZB, TPB), TPB, 0, stream>>>(scrF, v_b, BH * ZB);
    // logits[b,h] = (q_h @ k_h^T) / 8 (generic kernel: B column-major gather)
    gemm(stream, q_b, ZB, DH, FN, 1,
         k_b, ZB, DH, 1, FN,
         logit, (long)AH * ZB, ZB, EMB, 1,
         nullptr, 0, 0.125f, EMB, EMB, DH, BH * AH, AH);
    k_softmax_row<<<blks((long)BH * AH * EMB * 32, TPB), TPB, 0, stream>>>(
        logit, w_b, (long)BH * AH * EMB);
    // attn[b,h] = w @ v_h  (TDM: v panel row stride 256, tile 64 cols)
    gemmT(stream, w_b, (long)AH * ZB, ZB, EMB, 1,
          v_b, ZB, DH, FN, 64,
          scrF, ZB, DH, FN, 1,
          nullptr, 0, 1.f, EMB, DH, EMB, BH * AH, AH);
    k_f32_to_bf16<<<blks(BH * ZB, TPB), TPB, 0, stream>>>(scrF, at_b, BH * ZB);
    // recon = attn@Wo + bo -> d_out[0:]
    gemmT(stream, at_b, 0, 0, FN, 1, Wo_b, 0, 0, FN, FN, recon_out, 0, 0, FN, 1,
          bo, 0, 1.f, BH * EMB, FN, FN, 1, 1);

    // ---- MLP forecaster ----
    gemmT(stream, z_b, ZB, 0, 1, FN,
          W1_b, 0, 0, EMB, EMB,
          scrF, ZB, 0, EMB, 1,
          b1, 1, 1.f, FN, EMB, EMB, BH, 1);
    k_forecast<<<blks((long)BH * FN * 32, TPB), TPB, 0, stream>>>(
        scrF, W2, b2, fc_out, (long)BH * FN);
}
